// BPTBlock_66279935312382
// MI455X (gfx1250) — matrix operand order repaired and checked by reading, not compiled, not run on hardware
//
#include <hip/hip_runtime.h>
#include <hip/hip_bf16.h>

// Problem constants (match reference)
#define NN  50000
#define NE  1600000
#define DD  128
#define HH  8
#define DKK 16
#define FFD 512

typedef __attribute__((ext_vector_type(16))) __bf16 v16bf;
typedef __attribute__((ext_vector_type(8)))  __bf16 v8bf;
typedef __attribute__((ext_vector_type(8)))  float  v8f;

// ---------------------------------------------------------------- utilities

__global__ void f2bf_kernel(const float* __restrict__ x, __bf16* __restrict__ y, int n) {
  int i = blockIdx.x * blockDim.x + threadIdx.x;
  if (i < n) y[i] = (__bf16)x[i];
}

__global__ void fill_kernel(float* __restrict__ p, float v, int n) {
  int i = blockIdx.x * blockDim.x + threadIdx.x;
  if (i < n) p[i] = v;
}

__device__ inline void atomicMaxF(float* addr, float val) {
  unsigned int* ua = (unsigned int*)addr;
  unsigned int old = __float_as_uint(*addr);
  while (__uint_as_float(old) < val) {
    unsigned int assumed = old;
    old = atomicCAS(ua, assumed, __float_as_uint(val));
    if (old == assumed) break;
  }
}

// --------------------------------------------------- weight fragment packing
// Pack row-major W[K x Ncols] (fp32) into WMMA B-fragment order (bf16):
// tile (nt, kt) of 32x16; within tile, lane L holds 16 halves contiguously:
//   column = nt*16 + (L&15), K = kt*32 + (L>=16 ? 8 : 0) + (j&7) + (j>=8 ? 16 : 0)   (j = 0..15)
__global__ void pack_b_kernel(const float* __restrict__ W, __bf16* __restrict__ Bp,
                              int K, int Ncols) {
  int idx = blockIdx.x * blockDim.x + threadIdx.x;
  int total = K * Ncols;
  if (idx >= total) return;
  int KT   = K >> 5;
  int j    = idx & 15;
  int L    = (idx >> 4) & 31;
  int rest = idx >> 9;
  int kt   = rest % KT;
  int nt   = rest / KT;
  int kk   = kt * 32 + ((L >> 4) << 3) + (j & 7) + ((j >> 3) << 4);
  int c    = (nt << 4) + (L & 15);
  Bp[idx] = (__bf16)W[(size_t)kk * Ncols + c];
}

// --------------------------------------------------------------- WMMA GEMM
// C[M x Ncols] = A[M x KDIM](bf16, row-major) @ Bp(packed bf16) (+bias)(+relu)
// The 64-column B panel for this block is staged into LDS once via the CDNA5
// async-copy path (global_load_async_to_lds_b128 / ASYNCcnt), then all waves
// feed their WMMAs from ds_load. A fragments stream from global per wave.
template<int KDIM, bool BIAS, bool RELU, bool OUTBF>
__global__ void gemm_wmma_kernel(const __bf16* __restrict__ A,
                                 const __bf16* __restrict__ Bp,
                                 const float* __restrict__ bias,
                                 float* __restrict__ Cf,
                                 __bf16* __restrict__ Cb,
                                 int M, int Ncols) {
  constexpr int KT = KDIM / 32;
  // panel = 4 N-tiles x KT k-tiles x 512 bf16 (contiguous in packed layout)
  __shared__ __bf16 Bs[KT * 4 * 512];

  const int wave = threadIdx.x >> 5;
  const int lane = threadIdx.x & 31;
  const int mt = blockIdx.x * (blockDim.x >> 5) + wave;
  const bool active = (mt * 16 < M);        // wave-uniform

  // ---- cooperative async stage of the B panel into LDS (KT*4096 bytes) ----
  {
    const __bf16* gpanel = Bp + (size_t)(blockIdx.y * 4) * KT * 512;
    const unsigned lbase = (unsigned)(uintptr_t)&Bs[0]; // low 32b = LDS offset
    #pragma unroll
    for (int i = 0; i < KT; ++i) {
      const unsigned boff = (unsigned)((i * 256 + threadIdx.x) * 16);
      unsigned long long ga = (unsigned long long)(uintptr_t)gpanel + boff;
      unsigned la = lbase + boff;
      asm volatile("global_load_async_to_lds_b128 %0, %1, off"
                   :: "v"(la), "v"(ga) : "memory");
    }
    asm volatile("s_wait_asynccnt 0" ::: "memory");
  }
  __syncthreads();
  if (!active) return;                      // after barrier: no deadlock

  const int m0 = mt << 4;
  const int n0 = blockIdx.y << 6;
  const int r  = lane & 15;                 // A row / C column within tile
  const int hi = lane >> 4;                 // A K-chunk / C row-half select

  // A fragment source: row (m0+r), K chunks {hi*8 .. hi*8+7} and {+16..+23}
  const __bf16* arow = A + (size_t)(m0 + r) * KDIM + hi * 8;

  v8f acc[4] = {};
  #pragma unroll
  for (int kt = 0; kt < KT; ++kt) {
    v8bf alo = *(const v8bf*)(arow + kt * 32);
    v8bf ahi = *(const v8bf*)(arow + kt * 32 + 16);
    v16bf af = __builtin_shufflevector(alo, ahi,
        0,1,2,3,4,5,6,7,8,9,10,11,12,13,14,15);
    if (kt + 1 < KT) __builtin_prefetch(arow + (kt + 1) * 32, 0, 1);
    #pragma unroll
    for (int nn = 0; nn < 4; ++nn) {
      const v16bf bfr = *(const v16bf*)(&Bs[(((nn * KT + kt) * 32) + lane) * 16]);
      acc[nn] = __builtin_amdgcn_wmma_f32_16x16x32_bf16(
          false, af, false, bfr, (short)0, acc[nn], false, false);
    }
  }

  // C layout: lane L, vgpr v -> row m0 + (L>>4)*8 + v, col n0 + nn*16 + (L&15)
  const int rbase = m0 + hi * 8;
  #pragma unroll
  for (int nn = 0; nn < 4; ++nn) {
    const int col = n0 + nn * 16 + r;
    const float bv = BIAS ? bias[col] : 0.0f;
    #pragma unroll
    for (int v = 0; v < 8; ++v) {
      float val = acc[nn][v] + bv;
      if (RELU) val = fmaxf(val, 0.0f);
      const size_t o = (size_t)(rbase + v) * Ncols + col;
      if (OUTBF) Cb[o] = (__bf16)val;
      else       Cf[o] = val;
    }
  }
}

// --------------------------------------------------------- sparse attention
// qkv: [N x 384] fused, q at +0, k at +128, v at +256

__global__ void edge_score_kernel(const float* __restrict__ qkv,
                                  const int* __restrict__ src,
                                  const int* __restrict__ dst,
                                  float* __restrict__ e,
                                  float* __restrict__ mx) {
  int gid = blockIdx.x * blockDim.x + threadIdx.x;
  if (gid >= NE * HH) return;
  int m = gid >> 3, hd = gid & 7;
  int dv = dst[m], sv = src[m];
  const float4* q4 = (const float4*)(qkv + (size_t)dv * 384 + hd * DKK);
  const float4* k4 = (const float4*)(qkv + (size_t)sv * 384 + DD + hd * DKK);
  float s = 0.0f;
  #pragma unroll
  for (int i = 0; i < 4; ++i) {
    float4 qa = q4[i], ka = k4[i];
    s += qa.x * ka.x + qa.y * ka.y + qa.z * ka.z + qa.w * ka.w;
  }
  s *= 0.25f;                        // 1/sqrt(DK), DK=16
  e[gid] = s;
  atomicMaxF(&mx[dv * HH + hd], s);
}

__global__ void edge_exp_kernel(float* __restrict__ e,
                                const float* __restrict__ mx,
                                float* __restrict__ dn,
                                const int* __restrict__ dst) {
  int gid = blockIdx.x * blockDim.x + threadIdx.x;
  if (gid >= NE * HH) return;
  int m = gid >> 3, hd = gid & 7;
  int dv = dst[m];
  float ex = __expf(e[gid] - mx[dv * HH + hd]);
  e[gid] = ex;
  atomicAdd(&dn[dv * HH + hd], ex);
}

__global__ void edge_spmm_kernel(const float* __restrict__ e,
                                 const float* __restrict__ dn,
                                 const float* __restrict__ qkv,
                                 const int* __restrict__ src,
                                 const int* __restrict__ dst,
                                 float* __restrict__ a) {
  int gid = blockIdx.x * blockDim.x + threadIdx.x;
  if (gid >= NE * HH) return;
  int m = gid >> 3, hd = gid & 7;
  int dv = dst[m], sv = src[m];
  float al = e[gid] / dn[dv * HH + hd];
  const float* vr = qkv + (size_t)sv * 384 + 2 * DD + hd * DKK;
  float* ar = a + (size_t)dv * DD + hd * DKK;
  #pragma unroll
  for (int i = 0; i < DKK; ++i) atomicAdd(&ar[i], al * vr[i]);
}

// ------------------------------------------------------ residual + layernorm
// out = LN(x + r) * g + b ; one wave per row (wave32 -> 4 floats/lane)
__global__ void resid_ln_kernel(const float* __restrict__ x,
                                const float* __restrict__ r,
                                const float* __restrict__ g,
                                const float* __restrict__ b,
                                float* __restrict__ out,
                                __bf16* __restrict__ outb, int M) {
  int wave = threadIdx.x >> 5, lane = threadIdx.x & 31;
  int row = blockIdx.x * (blockDim.x >> 5) + wave;
  if (row >= M) return;
  const float4* x4 = (const float4*)(x + (size_t)row * DD);
  const float4* r4 = (const float4*)(r + (size_t)row * DD);
  float4 xa = x4[lane], ra = r4[lane];
  float v0 = xa.x + ra.x, v1 = xa.y + ra.y, v2 = xa.z + ra.z, v3 = xa.w + ra.w;
  float s = v0 + v1 + v2 + v3;
  #pragma unroll
  for (int off = 16; off > 0; off >>= 1) s += __shfl_xor(s, off, 32);
  float mu = s * (1.0f / DD);
  float d0 = v0 - mu, d1 = v1 - mu, d2 = v2 - mu, d3 = v3 - mu;
  float ss = d0 * d0 + d1 * d1 + d2 * d2 + d3 * d3;
  #pragma unroll
  for (int off = 16; off > 0; off >>= 1) ss += __shfl_xor(ss, off, 32);
  float rs = rsqrtf(ss * (1.0f / DD) + 1e-5f);
  int c = lane * 4;
  float o0 = d0 * rs * g[c + 0] + b[c + 0];
  float o1 = d1 * rs * g[c + 1] + b[c + 1];
  float o2 = d2 * rs * g[c + 2] + b[c + 2];
  float o3 = d3 * rs * g[c + 3] + b[c + 3];
  float4 ov; ov.x = o0; ov.y = o1; ov.z = o2; ov.w = o3;
  ((float4*)(out + (size_t)row * DD))[lane] = ov;
  if (outb) {
    __bf16* ob = outb + (size_t)row * DD + c;
    ob[0] = (__bf16)o0; ob[1] = (__bf16)o1; ob[2] = (__bf16)o2; ob[3] = (__bf16)o3;
  }
}

// --------------------------------------------------------------------- host

static inline dim3 blks(long long n, int t) { return dim3((unsigned)((n + t - 1) / t)); }

extern "C" void kernel_launch(void* const* d_in, const int* in_sizes, int n_in,
                              void* d_out, int out_size, void* d_ws, size_t ws_size,
                              hipStream_t stream) {
  const float* h     = (const float*)d_in[0];
  const int*   src   = (const int*)  d_in[1];
  const int*   dst   = (const int*)  d_in[2];
  const float* Wq    = (const float*)d_in[3];
  const float* Wk    = (const float*)d_in[4];
  const float* Wv    = (const float*)d_in[5];
  const float* Wo    = (const float*)d_in[6];
  const float* ln1g  = (const float*)d_in[7];
  const float* ln1b  = (const float*)d_in[8];
  const float* ln2g  = (const float*)d_in[9];
  const float* ln2b  = (const float*)d_in[10];
  const float* W1    = (const float*)d_in[11];
  const float* b1    = (const float*)d_in[12];
  const float* W2    = (const float*)d_in[13];
  const float* b2    = (const float*)d_in[14];
  float* out = (float*)d_out;

  // workspace carve-up (regions reused across phases; ~221 MB total)
  char* base = (char*)d_ws;
  size_t off = 0;
  auto alloc = [&](size_t bytes) -> void* {
    void* p = base + off;
    off += (bytes + 255) & ~(size_t)255;
    return p;
  };
  __bf16* bfA  = (__bf16*)alloc((size_t)NN * DD * 2);    // h_bf16 -> a_bf16 -> h1_bf16
  float*  qkv  = (float*) alloc((size_t)NN * 384 * 4);   // fused q|k|v
  float*  ebuf = (float*) alloc((size_t)NE * HH * 4);    // e/alpha, then f(bf16) N*FF
  float*  mx   = (float*) alloc((size_t)NN * HH * 4);
  float*  dn   = (float*) alloc((size_t)NN * HH * 4);
  float*  abuf = (float*) alloc((size_t)NN * DD * 4);    // attention aggregate
  float*  obuf = (float*) alloc((size_t)NN * DD * 4);    // o, then ffn
  float*  h1   = (float*) alloc((size_t)NN * DD * 4);
  __bf16* bq   = (__bf16*)alloc((size_t)DD * 384 * 2);   // packed Wq|Wk|Wv
  __bf16* bo   = (__bf16*)alloc((size_t)DD * DD * 2);
  __bf16* b1p  = (__bf16*)alloc((size_t)DD * FFD * 2);
  __bf16* b2p  = (__bf16*)alloc((size_t)FFD * DD * 2);
  __bf16* fbuf = (__bf16*)ebuf;                          // relu(h1@W1+b1), N*FF bf16

  const int MT = NN / 16;                    // 3125 M-tiles
  dim3 gemmBlk(256);                         // 8 waves per block
  unsigned gx = (unsigned)((MT + 7) / 8);

  // 1) h -> bf16
  f2bf_kernel<<<blks((long long)NN * DD, 256), 256, 0, stream>>>(h, bfA, NN * DD);

  // 2) pack all weights into WMMA fragment order
  pack_b_kernel<<<blks(DD * DD, 256), 256, 0, stream>>>(Wq, bq,                DD, DD);
  pack_b_kernel<<<blks(DD * DD, 256), 256, 0, stream>>>(Wk, bq + DD * DD,      DD, DD);
  pack_b_kernel<<<blks(DD * DD, 256), 256, 0, stream>>>(Wv, bq + 2 * DD * DD,  DD, DD);
  pack_b_kernel<<<blks(DD * DD,  256), 256, 0, stream>>>(Wo, bo,  DD, DD);
  pack_b_kernel<<<blks(DD * FFD, 256), 256, 0, stream>>>(W1, b1p, DD, FFD);
  pack_b_kernel<<<blks(FFD * DD, 256), 256, 0, stream>>>(W2, b2p, FFD, DD);

  // 3) init accumulators
  fill_kernel<<<blks((long long)NN * DD, 256), 256, 0, stream>>>(abuf, 0.0f, NN * DD);
  fill_kernel<<<blks(NN * HH, 256), 256, 0, stream>>>(dn, 0.0f, NN * HH);
  fill_kernel<<<blks(NN * HH, 256), 256, 0, stream>>>(mx, -3.0e38f, NN * HH);

  // 4) fused QKV projection: qkv = h_bf16 @ [Wq|Wk|Wv]
  gemm_wmma_kernel<128, false, false, false><<<dim3(gx, 384 / 64), gemmBlk, 0, stream>>>(
      bfA, bq, nullptr, qkv, (__bf16*)nullptr, NN, 384);

  // 5) sparse attention
  dim3 eg = blks((long long)NE * HH, 256);
  edge_score_kernel<<<eg, 256, 0, stream>>>(qkv, src, dst, ebuf, mx);
  edge_exp_kernel  <<<eg, 256, 0, stream>>>(ebuf, mx, dn, dst);
  edge_spmm_kernel <<<eg, 256, 0, stream>>>(ebuf, dn, qkv, src, dst, abuf);

  // 6) output projection: o = a_bf16 @ Wo
  f2bf_kernel<<<blks((long long)NN * DD, 256), 256, 0, stream>>>(abuf, bfA, NN * DD);
  gemm_wmma_kernel<128, false, false, false><<<dim3(gx, 128 / 64), gemmBlk, 0, stream>>>(
      bfA, bo, nullptr, obuf, (__bf16*)nullptr, NN, 128);

  // 7) h1 = LN(h + o)  (fp32 + bf16 copy for FFN)
  resid_ln_kernel<<<blks(NN, 8) /* 8 rows per 256-thr block */, 256, 0, stream>>>(
      h, obuf, ln1g, ln1b, h1, bfA, NN);

  // 8) FFN: f = relu(h1@W1 + b1) (bf16), ffn = f@W2 + b2
  gemm_wmma_kernel<128, true, true, true><<<dim3(gx, 512 / 64), gemmBlk, 0, stream>>>(
      bfA, b1p, b1, nullptr, fbuf, NN, 512);
  gemm_wmma_kernel<512, true, false, false><<<dim3(gx, 128 / 64), gemmBlk, 0, stream>>>(
      fbuf, b2p, b2, obuf, (__bf16*)nullptr, NN, 128);

  // 9) h2 = LN(h1 + ffn) -> d_out
  resid_ln_kernel<<<blks(NN, 8), 256, 0, stream>>>(
      h1, obuf, ln2g, ln2b, out, (__bf16*)nullptr, NN);
}
